// SparseMOE_58884001628580
// MI455X (gfx1250) — matrix operand order repaired and checked by reading, not compiled, not run on hardware
//
#include <hip/hip_runtime.h>

// Sparse top-2 MoE for MI455X (gfx1250, wave32).
// N=8192 tokens, D_in=D_out=2048, E=8 experts, top-2 gating.
// Math path: f16 WMMA (v_wmma_f32_16x16x32_f16) with f32 accumulation.
// Block tile: 64 rows x 256 cols, 8 waves as 2(m) x 4(n);
// each wave: 32x64 output = 8 x (16x16 f32 WMMA accumulators).

#define D_IN  2048
#define D_OUT 2048
#define NE    8
#define BUCKET_CAP 8192
#define SA_STRIDE 40   // 64 rows x 32 k (f16), padded: 80B rows, 16B aligned
#define SB_STRIDE 40   // 256 cols x 32 k (f16), K-contiguous per column

typedef __attribute__((ext_vector_type(16))) _Float16 v16h;
typedef __attribute__((ext_vector_type(8)))  _Float16 v8h;
typedef __attribute__((ext_vector_type(2)))  _Float16 v2h;
typedef __attribute__((ext_vector_type(8)))  float    v8f;

// ---------------------------------------------------------------- zero init
__global__ void moe_zero_kernel(float* __restrict__ out, int n,
                                int* __restrict__ counts) {
    int i = blockIdx.x * blockDim.x + threadIdx.x;
    if (i < n) out[i] = 0.0f;
    if (i < NE) counts[i] = 0;
}

// ---------------------------------------------------------------- gating
// One wave (32 lanes) per token: 8 dot products over D_in, shfl reduction,
// top-2 + 2-way softmax, bucket (token, weight) per expert via atomics.
__global__ __launch_bounds__(256)
void moe_gate_kernel(const float* __restrict__ x, const float* __restrict__ Wg,
                     const float* __restrict__ bg, int* __restrict__ counts,
                     int* __restrict__ btok, float* __restrict__ bw, int Ntok) {
    const int lane = threadIdx.x & 31;
    const int wv   = threadIdx.x >> 5;
    const int t    = blockIdx.x * 8 + wv;
    if (t >= Ntok) return;

    float acc[NE];
#pragma unroll
    for (int e = 0; e < NE; ++e) acc[e] = 0.0f;

    const float* xr = x + (size_t)t * D_IN;
    for (int d = lane; d < D_IN; d += 32) {
        const float xv = xr[d];
        const float4* wr = (const float4*)(Wg + (size_t)d * NE);
        const float4 wa = wr[0];
        const float4 wb = wr[1];
        acc[0] += xv * wa.x; acc[1] += xv * wa.y;
        acc[2] += xv * wa.z; acc[3] += xv * wa.w;
        acc[4] += xv * wb.x; acc[5] += xv * wb.y;
        acc[6] += xv * wb.z; acc[7] += xv * wb.w;
    }
#pragma unroll
    for (int e = 0; e < NE; ++e) {
#pragma unroll
        for (int off = 16; off > 0; off >>= 1)
            acc[e] += __shfl_xor(acc[e], off, 32);
    }

    if (lane == 0) {
        float v[NE];
#pragma unroll
        for (int e = 0; e < NE; ++e) v[e] = acc[e] + bg[e];
        // top-1 (strict > : first index wins ties, matching jax top_k)
        int i0 = 0; float b0 = v[0];
#pragma unroll
        for (int e = 1; e < NE; ++e) if (v[e] > b0) { b0 = v[e]; i0 = e; }
        // top-2
        int i1 = -1; float b1 = -3.402823e38f;
#pragma unroll
        for (int e = 0; e < NE; ++e)
            if (e != i0 && v[e] > b1) { b1 = v[e]; i1 = e; }
        // softmax over the two kept logits (others are -inf -> 0)
        const float ex  = expf(b1 - b0);
        const float inv = 1.0f / (1.0f + ex);
        const int s0 = atomicAdd(&counts[i0], 1);
        btok[i0 * BUCKET_CAP + s0] = t;
        bw  [i0 * BUCKET_CAP + s0] = inv;
        const int s1 = atomicAdd(&counts[i1], 1);
        btok[i1 * BUCKET_CAP + s1] = t;
        bw  [i1 * BUCKET_CAP + s1] = ex * inv;
    }
}

// ---------------------------------------------------------------- expert GEMM
// grid = (D_OUT/256 n-blocks, maxMtiles(64-row), NE experts), 256 threads.
// 8 waves as 2(m) x 4(n); wave = 32 rows x 64 cols = 8 WMMA accumulators.
__global__ __launch_bounds__(256)
void moe_expert_gemm_kernel(const float* __restrict__ x,
                            const float* __restrict__ We,
                            const float* __restrict__ be,
                            const int*   __restrict__ counts,
                            const int*   __restrict__ btok,
                            const float* __restrict__ bw,
                            float*       __restrict__ out) {
    const int e     = blockIdx.z;
    const int mtile = blockIdx.y;
    const int nblk  = blockIdx.x;
    const int cnt   = counts[e];
    if (mtile * 64 >= cnt) return;     // uniform early exit (no barriers yet)

    __shared__ _Float16 sA[64 * SA_STRIDE];    // 5 KB
    __shared__ _Float16 sB[256 * SB_STRIDE];   // 20 KB, K-contiguous per col
    __shared__ int   s_tok[64];
    __shared__ float s_w[64];

    const int tid = threadIdx.x;
    if (tid < 64) {
        const int slot = mtile * 64 + tid;
        const int cs   = (slot < cnt) ? slot : (cnt - 1);
        s_tok[tid] = btok[e * BUCKET_CAP + cs];
        s_w[tid]   = (slot < cnt) ? bw[e * BUCKET_CAP + cs] : 0.0f; // w=0 pads
    }

    const int lane = tid & 31;
    const int wv   = tid >> 5;
    const int mw   = wv >> 2;     // 0..1  (wave row group: 32 rows each)
    const int nw   = wv & 3;      // 0..3  (wave col group: 64 cols each)
    const int nl   = lane & 15;   // N (and A-row) within 16x16 fragment
    const int kh   = lane >> 4;   // half-wave K-block selector

    const float* Bbase = We + (size_t)e * D_IN * D_OUT + nblk * 256;

    // A staging coords: row = tid>>2 (0..63), col octet = (tid&3)*8
    const int arow = tid >> 2;
    const int acol = (tid & 3) * 8;
    // B staging coords: column pair bn, k-pair selector bk2
    const int bn   = (tid & 127) * 2;
    const int bk2  = (tid >> 7) * 2;

    v8f c[2][4];
#pragma unroll
    for (int g = 0; g < 2; ++g)
#pragma unroll
        for (int f = 0; f < 4; ++f)
            c[g][f] = (v8f){0.f, 0.f, 0.f, 0.f, 0.f, 0.f, 0.f, 0.f};

    for (int k0 = 0; k0 < D_IN; k0 += 32) {
        __syncthreads();   // also covers s_tok/s_w on first iteration

        // ---- stage A tile: 64 gathered token rows x 32 k, f32 -> f16 packed,
        //      one aligned 16B LDS store per thread
        {
            const float* xp = x + (size_t)s_tok[arow] * D_IN + k0 + acol;
            const float4 v0 = *(const float4*)xp;
            const float4 v1 = *(const float4*)(xp + 4);
            v8h h;
            h[0] = (_Float16)v0.x; h[1] = (_Float16)v0.y;
            h[2] = (_Float16)v0.z; h[3] = (_Float16)v0.w;
            h[4] = (_Float16)v1.x; h[5] = (_Float16)v1.y;
            h[6] = (_Float16)v1.z; h[7] = (_Float16)v1.w;
            *(v8h*)(&sA[arow * SA_STRIDE + acol]) = h;
        }
        // ---- stage B tile: 32 k x 256 n, transposed to K-contiguous f16.
        // Each pass: two float2 from adjacent k rows (coalesced across lanes),
        // pack column-wise -> two b32 LDS stores.
#pragma unroll
        for (int p = 0; p < 8; ++p) {
            const int k = p * 4 + bk2;
            const float2 r0 = *(const float2*)(Bbase
                                + (size_t)(k0 + k) * D_OUT + bn);
            const float2 r1 = *(const float2*)(Bbase
                                + (size_t)(k0 + k + 1) * D_OUT + bn);
            v2h p0; p0[0] = (_Float16)r0.x; p0[1] = (_Float16)r1.x;
            v2h p1; p1[0] = (_Float16)r0.y; p1[1] = (_Float16)r1.y;
            *(v2h*)(&sB[(bn + 0) * SB_STRIDE + k]) = p0;
            *(v2h*)(&sB[(bn + 1) * SB_STRIDE + k]) = p1;
        }
        __syncthreads();

        // ---- 2 A fragments (row groups g=0,1): lane holds M=nl,
        //      K = kh*8+{0..7} in elems 0..7, K = 16+kh*8+{0..7} in 8..15
        v16h a[2];
#pragma unroll
        for (int g = 0; g < 2; ++g) {
            const _Float16* ap =
                &sA[(mw * 32 + g * 16 + nl) * SA_STRIDE + kh * 8];
            const v8h alo = *(const v8h*)ap;
            const v8h ahi = *(const v8h*)(ap + 16);
            a[g] = __builtin_shufflevector(alo, ahi,
                0,1,2,3,4,5,6,7,8,9,10,11,12,13,14,15);
        }

        // ---- 4 B fragments (transient), each used by both A row groups
#pragma unroll
        for (int f = 0; f < 4; ++f) {
            const _Float16* bp =
                &sB[(nw * 64 + f * 16 + nl) * SB_STRIDE + kh * 8];
            const v8h blo = *(const v8h*)bp;
            const v8h bhi = *(const v8h*)(bp + 16);
            const v16h b = __builtin_shufflevector(blo, bhi,
                0,1,2,3,4,5,6,7,8,9,10,11,12,13,14,15);
            c[0][f] = __builtin_amdgcn_wmma_f32_16x16x32_f16(
                          false, a[0], false, b, (short)0, c[0][f],
                          false, false);
            c[1][f] = __builtin_amdgcn_wmma_f32_16x16x32_f16(
                          false, a[1], false, b, (short)0, c[1][f],
                          false, false);
        }
    }

    // ---- epilogue: out[tok] += w * (acc + bias); exactly 2 adds per element
#pragma unroll
    for (int f = 0; f < 4; ++f) {
        const int n = nblk * 256 + nw * 64 + f * 16 + nl;
        const float bias = be[e * D_OUT + n];
#pragma unroll
        for (int g = 0; g < 2; ++g) {
#pragma unroll
            for (int r = 0; r < 8; ++r) {
                const int m = mw * 32 + g * 16 + r + 8 * kh;
                const float w = s_w[m];
                atomicAdd(out + (size_t)s_tok[m] * D_OUT + n,
                          (c[g][f][r] + bias) * w);
            }
        }
    }
}

// ---------------------------------------------------------------- launcher
extern "C" void kernel_launch(void* const* d_in, const int* in_sizes, int n_in,
                              void* d_out, int out_size, void* d_ws,
                              size_t ws_size, hipStream_t stream) {
    const float* x  = (const float*)d_in[0];
    const float* We = (const float*)d_in[1];
    const float* be = (const float*)d_in[2];
    const float* Wg = (const float*)d_in[3];
    const float* bg = (const float*)d_in[4];
    float* out = (float*)d_out;

    const int Ntok = in_sizes[0] / D_IN;   // 8192

    // workspace: [counts: 8 ints][pad to 64B][btok: 8x8192 int][bw: 8x8192 f32]
    int*   counts = (int*)d_ws;
    int*   btok   = (int*)((char*)d_ws + 64);
    float* bw     = (float*)((char*)d_ws + 64
                             + (size_t)NE * BUCKET_CAP * sizeof(int));

    moe_zero_kernel<<<(out_size + 255) / 256, 256, 0, stream>>>(
        out, out_size, counts);

    moe_gate_kernel<<<(Ntok + 7) / 8, 256, 0, stream>>>(
        x, Wg, bg, counts, btok, bw, Ntok);

    dim3 grid(D_OUT / 256, (Ntok + 63) / 64, NE);
    moe_expert_gemm_kernel<<<grid, 256, 0, stream>>>(
        x, We, be, counts, btok, bw, out);
}
